// WNO1d_65927747994060
// MI455X (gfx1250) — matrix operand order repaired and checked by reading, not compiled, not run on hardware
//
#include <hip/hip_runtime.h>

typedef __attribute__((ext_vector_type(16))) _Float16 v16h;
typedef __attribute__((ext_vector_type(8)))  float    v8f;

#define NB   128
#define NC   64
#define NS   1024
#define NBC  8192    // NB*NC

// db6 decomposition filters (forward order). dwt uses [11-k] (pre-reversed
// correlation == true convolution); idwt (transposed conv) uses [k].
__device__ __constant__ float cDEC_LO[12] = {
  -1.07730108499558e-03f,  4.777257511010651e-03f,  5.538422009938016e-04f,
  -3.1582039318031156e-02f, 2.752286553001629e-02f,  9.750160558707936e-02f,
  -1.2976686756709563e-01f,-2.2626469396516913e-01f, 3.152503517092432e-01f,
   7.511339080215775e-01f,  4.946238903983854e-01f,  1.1154074335008017e-01f };
__device__ __constant__ float cDEC_HI[12] = {
  -1.1154074335008017e-01f, 4.946238903983854e-01f, -7.511339080215775e-01f,
   3.152503517092432e-01f,  2.2626469396516913e-01f,-1.2976686756709563e-01f,
  -9.750160558707936e-02f,  2.752286553001629e-02f,  3.1582039318031156e-02f,
   5.538422009938016e-04f, -4.777257511010651e-03f, -1.07730108499558e-03f };

__device__ __forceinline__ float gelu_f(float x) {
  return 0.5f * x * (1.0f + erff(x * 0.7071067811865475f));
}

// ---------------- fc0: h[b,c,s] = x[b,s]*w[0,c] + grid(s)*w[1,c] + bias[c]
__global__ __launch_bounds__(256) void fc0_k(
    const float* __restrict__ x, const float* __restrict__ w,
    const float* __restrict__ bias, float* __restrict__ h) {
  long t = (long)blockIdx.x * 256 + threadIdx.x;
  if (t >= (long)NB * NC * NS) return;
  int  s = (int)(t & (NS - 1));
  int  c = (int)((t >> 10) & (NC - 1));
  long b = t >> 16;
  float xv = x[(b << 10) + s];
  float g  = (float)s * (1.0f / (float)(NS - 1));
  h[t] = fmaf(xv, w[c], fmaf(g, w[NC + c], bias[c]));
}

// ---------------- forward DWT step (symmetric pad, stride-2, 12-tap)
__global__ __launch_bounds__(256) void dwt_k(
    const float* __restrict__ in, int ins, int Nin,
    float* __restrict__ lo, int los,
    float* __restrict__ hi, int Nout) {
  long t = (long)blockIdx.x * 256 + threadIdx.x;
  if (t >= (long)NBC * Nout) return;
  int  j  = (int)(t % Nout);
  long bc = t / Nout;
  int  pl = (2 * (Nout - 1) - Nin + 12) >> 1;
  const float* src = in + bc * (long)ins;
  float al = 0.f, ah = 0.f;
#pragma unroll
  for (int k = 0; k < 12; ++k) {
    int idx = 2 * j + k - pl;
    while (idx < 0 || idx >= Nin) {          // symmetric reflection
      if (idx < 0)    idx = -idx - 1;
      if (idx >= Nin) idx = 2 * Nin - 1 - idx;
    }
    float v = src[idx];
    al = fmaf(v, cDEC_LO[11 - k], al);
    ah = fmaf(v, cDEC_HI[11 - k], ah);
  }
  lo[bc * (long)los + j]  = al;
  hi[bc * (long)Nout + j] = ah;
}

// ---------------- inverse DWT step (lhs_dilation=2, pad(1,1), 12-tap corr.)
__global__ __launch_bounds__(256) void idwt_k(
    const float* __restrict__ lo, int los,
    const float* __restrict__ hi, int his, int N,
    float* __restrict__ out, int outs, int Nout) {
  long t = (long)blockIdx.x * 256 + threadIdx.x;
  if (t >= (long)NBC * Nout) return;
  int  j  = (int)(t % Nout);
  long bc = t / Nout;
  const float* L = lo + bc * (long)los;
  const float* H = hi + bc * (long)his;
  float acc = 0.f;
#pragma unroll
  for (int k = 0; k < 12; ++k) {
    int tt = j + k;
    if (tt & 1) {
      int i = tt >> 1;
      if (i < N) acc = fmaf(L[i], cDEC_LO[k], fmaf(H[i], cDEC_HI[k], acc));
    }
  }
  out[bc * (long)outs + j] = acc;
}

// ---------------- final inverse step fused with +x2 (+ optional gelu) -> h
__global__ __launch_bounds__(256) void idwt_final_k(
    const float* __restrict__ lo, int los,
    const float* __restrict__ hi, int his, int N,
    const float* __restrict__ x2, float* __restrict__ outh, int do_gelu) {
  long t = (long)blockIdx.x * 256 + threadIdx.x;
  if (t >= (long)NBC * NS) return;
  int  j  = (int)(t & (NS - 1));
  long bc = t >> 10;
  const float* L = lo + bc * (long)los;
  const float* H = hi + bc * (long)his;
  float acc = 0.f;
#pragma unroll
  for (int k = 0; k < 12; ++k) {
    int tt = j + k;
    if (tt & 1) {
      int i = tt >> 1;
      if (i < N) acc = fmaf(L[i], cDEC_LO[k], fmaf(H[i], cDEC_HI[k], acc));
    }
  }
  float v = acc + x2[t];
  if (do_gelu) v = gelu_f(v);
  outh[t] = v;
}

// ---------------- bottom-level channel mixes: 'bix,iox->box' at 14 positions
__global__ __launch_bounds__(256) void einsum_k(
    const float* __restrict__ loin, int los,
    const float* __restrict__ hiin, int his,
    const float* __restrict__ w1, const float* __restrict__ w2,
    float* __restrict__ loE, float* __restrict__ hiE) {
  int t = blockIdx.x * 256 + threadIdx.x;
  if (t >= NB * NC * 14) return;
  int xx = t % 14;
  int o  = (t / 14) & 63;
  int b  = t / (14 * 64);
  float al = 0.f, ah = 0.f;
  for (int i = 0; i < 64; ++i) {
    al = fmaf(loin[(long)(b * 64 + i) * los + xx], w1[(i * 64 + o) * 14 + xx], al);
    ah = fmaf(hiin[(long)(b * 64 + i) * his + xx], w2[(i * 64 + o) * 14 + xx], ah);
  }
  loE[(long)(b * 64 + o) * 14 + xx] = al;
  hiE[(long)(b * 64 + o) * 14 + xx] = ah;
}

// ---------------- x2 = conv_w(64x64) @ h(64xS) per batch, via WMMA f16->f32
// One wave per 16(o) x 16(s) output tile; K=64 in two 16x16x32 WMMAs.
__global__ __launch_bounds__(256) void conv_x2_wmma(
    const float* __restrict__ h, const float* __restrict__ W,
    const float* __restrict__ bias, float* __restrict__ out) {
  int wave = threadIdx.x >> 5, lane = threadIdx.x & 31;
  long tile = (long)blockIdx.x * 8 + wave;      // 128*4*64 = 32768 tiles exact
  int st = (int)(tile & 63);
  int ot = (int)((tile >> 6) & 3);
  long b = tile >> 8;
  int m  = lane & 15, hb = lane >> 4;
  const float* hB = h + (b << 16);
  v8f c = {};
#pragma unroll
  for (int ks = 0; ks < 64; ks += 32) {
    v16h a, bm;
#pragma unroll
    for (int j = 0; j < 16; ++j) {              // A: 16x32, rows=o, K=i
      int k = ((j < 8) ? j : j + 8) + hb * 8 + ks;
      a[j] = (_Float16)W[(ot * 16 + m) * 64 + k];
    }
#pragma unroll
    for (int j = 0; j < 16; ++j) {              // B: 32x16, K=i, cols=s
      int k = ((j < 8) ? j : j + 8) + hb * 8 + ks;
      bm[j] = (_Float16)hB[k * 1024 + st * 16 + m];
    }
    c = __builtin_amdgcn_wmma_f32_16x16x32_f16(false, a, false, bm,
                                               (short)0, c, false, false);
  }
#pragma unroll
  for (int r = 0; r < 8; ++r) {
    int o = ot * 16 + r + hb * 8;
    out[(b << 16) + o * 1024 + st * 16 + m] = c[r] + bias[o];
  }
}

// ---------------- fused head: out = gelu(h^T @ fc1 + b1) @ fc2 + b2
// One wave: 16 rows (b,s) x all 128 hidden cols via 16 WMMAs, then in-wave
// gelu + fc2 dot + shfl_xor reduction (no 64MB intermediate).
__global__ __launch_bounds__(256) void head_wmma(
    const float* __restrict__ h,  const float* __restrict__ w1,
    const float* __restrict__ b1, const float* __restrict__ w2,
    const float* __restrict__ b2, float* __restrict__ out) {
  int wave = threadIdx.x >> 5, lane = threadIdx.x & 31;
  int m = lane & 15, hb = lane >> 4;
  long tile = (long)blockIdx.x * 8 + wave;      // 131072/16 = 8192 tiles exact
  long p0 = tile * 16;
  long p  = p0 + m;
  const float* hrow = h + ((p >> 10) << 16) + (p & 1023);   // + k*1024
  v8f zero = {};
  v8f c[8];
#pragma unroll
  for (int nt = 0; nt < 8; ++nt) c[nt] = zero;
#pragma unroll
  for (int ks = 0; ks < 64; ks += 32) {
    v16h a;
#pragma unroll
    for (int j = 0; j < 16; ++j) {              // A: rows=(b,s), K=c (from h^T)
      int k = ((j < 8) ? j : j + 8) + hb * 8 + ks;
      a[j] = (_Float16)hrow[k * 1024];
    }
#pragma unroll
    for (int nt = 0; nt < 8; ++nt) {            // B: fc1_w (64x128) col tiles
      v16h bm;
#pragma unroll
      for (int j = 0; j < 16; ++j) {
        int k = ((j < 8) ? j : j + 8) + hb * 8 + ks;
        bm[j] = (_Float16)w1[k * 128 + nt * 16 + m];
      }
      c[nt] = __builtin_amdgcn_wmma_f32_16x16x32_f16(false, a, false, bm,
                                                     (short)0, c[nt], false, false);
    }
  }
  float r[8];
#pragma unroll
  for (int e = 0; e < 8; ++e) r[e] = 0.f;
#pragma unroll
  for (int nt = 0; nt < 8; ++nt) {
    int col = nt * 16 + m;
    float wc = w2[col], bc = b1[col];
#pragma unroll
    for (int e = 0; e < 8; ++e)
      r[e] = fmaf(gelu_f(c[nt][e] + bc), wc, r[e]);
  }
#pragma unroll
  for (int e = 0; e < 8; ++e) {                 // reduce within each 16-lane half
    r[e] += __shfl_xor(r[e], 8, 32);
    r[e] += __shfl_xor(r[e], 4, 32);
    r[e] += __shfl_xor(r[e], 2, 32);
    r[e] += __shfl_xor(r[e], 1, 32);
  }
  if (m == 0) {
    float bb = b2[0];
#pragma unroll
    for (int e = 0; e < 8; ++e)
      out[p0 + e + hb * 8] = r[e] + bb;
  }
}

static inline int nblk(long n) { return (int)((n + 255) / 256); }

extern "C" void kernel_launch(void* const* d_in, const int* in_sizes, int n_in,
                              void* d_out, int out_size, void* d_ws, size_t ws_size,
                              hipStream_t stream) {
  (void)in_sizes; (void)n_in; (void)out_size; (void)ws_size;
  const float* x      = (const float*)d_in[0];
  const float* fc0_w  = (const float*)d_in[1];
  const float* fc0_b  = (const float*)d_in[2];
  const float* wc_w1  = (const float*)d_in[3];
  const float* wc_w2  = (const float*)d_in[4];
  const float* conv_w = (const float*)d_in[5];
  const float* conv_b = (const float*)d_in[6];
  const float* fc1_w  = (const float*)d_in[7];
  const float* fc1_b  = (const float*)d_in[8];
  const float* fc2_w  = (const float*)d_in[9];
  const float* fc2_b  = (const float*)d_in[10];
  float* out = (float*)d_out;
  float* ws  = (float*)d_ws;

  // Workspace (floats). Region R is a union: during the wavelet cascade it
  // holds {loB | hi pyramid levels 2..8 | loE | hiE}; after inverse step 7 it
  // is dead and reused as the x2 GEMM output. Total ~96.3 MiB.
  float* bufH = ws;                                   // 8,388,608
  float* hi1  = bufH + (size_t)NBC * NS;              // 517*8192
  float* loA  = hi1  + (size_t)517 * NBC;             // 518*8192
  float* R    = loA  + (size_t)518 * NBC;             // 8,388,608 (union)
  float* loB  = R;                                    // 264*8192
  float* pyr2 = R    + (size_t)264 * NBC;             // 575*8192 (levels 2..8)
  float* loE  = pyr2 + (size_t)575 * NBC;             // 14*8192
  float* hiE  = loE  + (size_t)14  * NBC;             // 14*8192
  float* x2b  = R;                                    // alias (after cascade)

  static const int Ns[9]     = {1024, 517, 264, 137, 74, 42, 26, 18, 14};
  static const int pyrOff[7] = {0, 264, 401, 475, 517, 543, 561};

  fc0_k<<<nblk((long)NBC * NS), 256, 0, stream>>>(x, fc0_w, fc0_b, bufH);

  for (int layer = 0; layer < 4; ++layer) {
    const float* Wl  = conv_w + layer * 4096;
    const float* bl  = conv_b + layer * 64;
    const float* w1l = wc_w1 + (size_t)layer * 64 * 64 * 14;
    const float* w2l = wc_w2 + (size_t)layer * 64 * 64 * 14;

    // ---- forward DWT cascade (8 levels), lo ping-pongs loA/loB
    for (int l = 0; l < 8; ++l) {
      int Nin = Ns[l], Nout = Ns[l + 1];
      const float* in; int ins;
      if (l == 0)      { in = bufH; ins = 1024; }
      else if (l & 1)  { in = loA;  ins = 518;  }
      else             { in = loB;  ins = 264;  }
      float* lo; int los;
      if ((l & 1) == 0) { lo = loA; los = 518; } else { lo = loB; los = 264; }
      float* hi = (l == 0) ? hi1 : (pyr2 + (size_t)pyrOff[l - 1] * NBC);
      dwt_k<<<nblk((long)NBC * Nout), 256, 0, stream>>>(in, ins, Nin, lo, los, hi, Nout);
    }

    // ---- bottom channel mixes (lo in loB len 14; hi level-8 in pyramid)
    einsum_k<<<nblk((long)NB * NC * 14), 256, 0, stream>>>(
        loB, 264, pyr2 + (size_t)561 * NBC, 14, w1l, w2l, loE, hiE);

    // ---- inverse DWT, steps 1..7 (out lens 18,26,42,74,138,264,518)
    idwt_k<<<nblk((long)NBC *  18), 256, 0, stream>>>(loE, 14,  hiE, 14, 14,  loA, 518,  18);
    idwt_k<<<nblk((long)NBC *  26), 256, 0, stream>>>(loA, 518, pyr2 + (size_t)543 * NBC, 18,  18,  loB, 264,  26);
    idwt_k<<<nblk((long)NBC *  42), 256, 0, stream>>>(loB, 264, pyr2 + (size_t)517 * NBC, 26,  26,  loA, 518,  42);
    idwt_k<<<nblk((long)NBC *  74), 256, 0, stream>>>(loA, 518, pyr2 + (size_t)475 * NBC, 42,  42,  loB, 264,  74);
    idwt_k<<<nblk((long)NBC * 138), 256, 0, stream>>>(loB, 264, pyr2 + (size_t)401 * NBC, 74,  74,  loA, 518, 138);
    idwt_k<<<nblk((long)NBC * 264), 256, 0, stream>>>(loA, 518, pyr2 + (size_t)264 * NBC, 137, 137, loB, 264, 264);
    idwt_k<<<nblk((long)NBC * 518), 256, 0, stream>>>(loB, 264, pyr2, 264, 264, loA, 518, 518);

    // ---- x2 = conv_w @ h (+bias) via WMMA; R is now reusable, h is intact
    conv_x2_wmma<<<4096, 256, 0, stream>>>(bufH, Wl, bl, x2b);

    // ---- final inverse step + x2 (+ gelu for layers 0..2) -> new h
    idwt_final_k<<<nblk((long)NBC * NS), 256, 0, stream>>>(
        loA, 518, hi1, 517, 517, x2b, bufH, (layer < 3) ? 1 : 0);
  }

  // ---- fused head: fc1 (WMMA) -> gelu -> fc2, straight to d_out
  head_wmma<<<1024, 256, 0, stream>>>(bufH, fc1_w, fc1_b, fc2_w, fc2_b, out);
}
